// CustomRNNLayer_80272938762505
// MI455X (gfx1250) — matrix operand order repaired and checked by reading, not compiled, not run on hardware
//
#include <hip/hip_runtime.h>
#include <hip/hip_bf16.h>
#include <stdint.h>

// Elman RNN, T=4096, D=2048:  h_t = tanh(x_t @ Wxh + h_{t-1} @ Whh)
// Phase 1: A = XS @ Wxh, bf16 WMMA GEMM, double-buffered LDS staging of B tiles.
// Phase 2: persistent-kernel sequential recurrence; Whh bf16 from L2; h_{t-1}
//          staged to LDS via GLOBAL_LOAD_ASYNC_TO_LDS_B128 (ASYNCcnt path);
//          device-wide software barrier between steps.

#define T_STEPS 4096
#define DIM     2048
#define NBLK    64          // persistent workgroups for the recurrence phase
#define BSTRIDE 40          // LDS row stride in bf16 (80 B = 20 dwords, bank-spread)

typedef __attribute__((ext_vector_type(16))) __bf16 v16bf;
typedef __attribute__((ext_vector_type(8)))  __bf16 v8bf;
typedef __attribute__((ext_vector_type(8)))  float  v8f;

// ---------------------------------------------------------------- barrier init
__global__ void init_barrier_kernel(unsigned int* bar) {
    if (threadIdx.x == 0) *bar = 0u;
}

// ---------------------------------------------------------------- fp32 -> bf16
__global__ void convert_bf16_kernel(const float* __restrict__ xs,
                                    const float* __restrict__ wxh,
                                    const float* __restrict__ whh,
                                    __bf16* __restrict__ xsb,
                                    __bf16* __restrict__ wxb,
                                    __bf16* __restrict__ whb) {
    const int n_xs = T_STEPS * DIM;
    const int n_w  = DIM * DIM;
    const int stride = gridDim.x * blockDim.x;
    for (int i = blockIdx.x * blockDim.x + threadIdx.x; i < n_xs; i += stride)
        xsb[i] = (__bf16)xs[i];
    for (int i = blockIdx.x * blockDim.x + threadIdx.x; i < n_w; i += stride) {
        wxb[i] = (__bf16)wxh[i];
        whb[i] = (__bf16)whh[i];
    }
}

// ------------------------------------------------- Phase 1: A = XS @ Wxh (WMMA)
// Block = 256 threads = 8 waves, computing a 16(M) x 128(N) strip.
// Double-buffered LDS: while tile k is consumed (ds_load_b128 -> v_wmma),
// tile k+1's coalesced global loads are already in flight; its transposed
// LDS stores land in the alternate buffer after the WMMA.
__global__ void __launch_bounds__(256)
gemm_xs_wxh_wmma(const __bf16* __restrict__ xsb,
                 const __bf16* __restrict__ wxb,
                 float* __restrict__ A) {
    __shared__ __bf16 lb[2][128 * BSTRIDE];       // 2 x 10 KB

    const int tid  = threadIdx.x;
    const int wave = tid >> 5;
    const int lid  = tid & 31;

    const int nblks = DIM / 128;                  // 16
    const int trow  = blockIdx.x / nblks;         // 0..255 (M tile)
    const int n0    = (blockIdx.x % nblks) * 128; // N strip base

    const int m      = trow * 16 + (lid & 15);    // A row this lane supplies
    const int a_koff = (lid < 16) ? 0 : 8;        // ISA 16-bit A 16x32 layout
    const int nloc   = wave * 16 + (lid & 15);    // local N column (0..127)
    const int b_k0   = (lid < 16) ? 0 : 8;        // ISA 16-bit B 32x16 layout

    // staging map: thread -> (k row, 16-column group) of each 32x128 tile
    const int st_k  = tid >> 3;                   // 0..31
    const int st_ng = (tid & 7) * 16;             // 0,16,...,112
    const __bf16* bsrc = wxb + (size_t)st_k * DIM + n0 + st_ng;

    // ---- preload tile 0 into buffer 0
    {
        v8bf r0 = *(const v8bf*)(bsrc);
        v8bf r1 = *(const v8bf*)(bsrc + 8);
#pragma unroll
        for (int j = 0; j < 8; ++j) lb[0][(st_ng + j) * BSTRIDE + st_k] = r0[j];
#pragma unroll
        for (int j = 0; j < 8; ++j) lb[0][(st_ng + 8 + j) * BSTRIDE + st_k] = r1[j];
    }

    v8f acc = {};
    const int NIT = DIM / 32;                     // 64 K-steps
    for (int it = 0; it < NIT; ++it) {
        __syncthreads();                          // tile `it` staged & prior reads done

        // issue next tile's global loads early (overlap with WMMA below)
        v8bf nx0, nx1;
        if (it + 1 < NIT) {
            const __bf16* src = bsrc + (size_t)(it + 1) * 32 * DIM;
            nx0 = *(const v8bf*)(src);
            nx1 = *(const v8bf*)(src + 8);
        }

        const int kb = it * 32;
        // A fragment: two contiguous 16B global loads per lane
        const __bf16* ap = xsb + (size_t)m * DIM + kb + a_koff;
        v8bf alo = *(const v8bf*)(ap);
        v8bf ahi = *(const v8bf*)(ap + 16);
        v16bf afrag;
#pragma unroll
        for (int e = 0; e < 8; ++e) { afrag[e] = alo[e]; afrag[8 + e] = ahi[e]; }

        // B fragment: two contiguous 16B LDS loads per lane (transposed tile)
        const __bf16* bl = &lb[it & 1][nloc * BSTRIDE + b_k0];
        v8bf blo = *(const v8bf*)(bl);
        v8bf bhi = *(const v8bf*)(bl + 16);
        v16bf bfrag;
#pragma unroll
        for (int e = 0; e < 8; ++e) { bfrag[e] = blo[e]; bfrag[8 + e] = bhi[e]; }

        acc = __builtin_amdgcn_wmma_f32_16x16x32_bf16(
            /*neg_a=*/false, afrag, /*neg_b=*/false, bfrag,
            /*c_mod=*/(short)0, acc, /*reuse_a=*/false, /*reuse_b=*/false);

        // stash next tile (transposed) into the alternate buffer
        if (it + 1 < NIT) {
            __bf16* dst = lb[(it + 1) & 1];
#pragma unroll
            for (int j = 0; j < 8; ++j) dst[(st_ng + j) * BSTRIDE + st_k] = nx0[j];
#pragma unroll
            for (int j = 0; j < 8; ++j) dst[(st_ng + 8 + j) * BSTRIDE + st_k] = nx1[j];
        }
    }

    // C/D layout: VGPR r -> M = r + 8*(lane>=16), N = lane%16
    const int mrow = trow * 16 + (lid >> 4) * 8;
#pragma unroll
    for (int r = 0; r < 8; ++r)
        A[(size_t)(mrow + r) * DIM + n0 + nloc] = acc[r];
}

// --------------------------------------- Phase 2: sequential recurrence (GEMV)
// 64 blocks x 256 threads, persistent. Each block owns 16 column-pairs (32 cols).
// h_{t-1} is staged once per block per step into LDS with async global->LDS
// copies (ASYNCcnt), removing the 16x redundant L2 reads of h.
__global__ void __launch_bounds__(256)
rnn_recurrence(const float* __restrict__ A,
               const __bf16* __restrict__ whb,
               float* __restrict__ out,
               unsigned int* __restrict__ bar) {
    __shared__ __align__(16) float hbuf[DIM];             // 8 KB: h_{t-1}
    __shared__ float red[256 * 2];
    const int tid        = threadIdx.x;
    const int pair_local = tid >> 4;                      // 0..15
    const int seg        = tid & 15;                      // 0..15
    const int cp         = blockIdx.x * 16 + pair_local;  // 0..1023
    const int j0         = cp * 2;                        // output columns j0, j0+1
    const int i0         = seg * (DIM / 16);              // 128-row reduction window

    // this thread's fixed slice of the h row (16 B x 2)
    const uint32_t hlds = (uint32_t)(uintptr_t)(&hbuf[tid * 8]);

    unsigned int target = 0;
    for (int t = 0; t < T_STEPS; ++t) {
        if (t > 0) {
            // async copy h_{t-1} (8 KB/block) from global to LDS; each thread 32 B
            const float* gsrc = out + (size_t)(t - 1) * DIM + tid * 8;
            asm volatile(
                "global_load_async_to_lds_b128 %0, %1, off\n\t"
                "global_load_async_to_lds_b128 %0, %1, off offset:16\n\t"
                "s_wait_asynccnt 0x0"
                :: "v"(hlds), "v"((uint64_t)(uintptr_t)gsrc)
                : "memory");
        }
        // pull next step's (single-use) bias row toward us early
        if (seg == 0 && t + 1 < T_STEPS)
            __builtin_prefetch(&A[(size_t)(t + 1) * DIM + j0], 0, 1);
        __syncthreads();                                  // hbuf visible block-wide

        float s0 = 0.f, s1 = 0.f;
        if (t > 0) {
#pragma unroll 4
            for (int ii = 0; ii < DIM / 16; ++ii) {
                // rotate start by 4*seg: LDS banks (ii+4*seg)%64 distinct per seg
                const int i = i0 + ((ii + 4 * seg) & (DIM / 16 - 1));
                float h = hbuf[i];
                // packed bf16 pair Whh[i][j0], Whh[i][j0+1] (resident in L2)
                uint32_t p = *(const uint32_t*)(whb + (size_t)i * DIM + j0);
                float w0 = __uint_as_float(p << 16);
                float w1 = __uint_as_float(p & 0xffff0000u);
                s0 = fmaf(h, w0, s0);
                s1 = fmaf(h, w1, s1);
            }
        }
        red[tid * 2]     = s0;
        red[tid * 2 + 1] = s1;
        __syncthreads();

        if (seg == 0) {
            float t0 = 0.f, t1 = 0.f;
#pragma unroll
            for (int u = 0; u < 16; ++u) {
                t0 += red[(pair_local * 16 + u) * 2];
                t1 += red[(pair_local * 16 + u) * 2 + 1];
            }
            // A row is read exactly once ever -> non-temporal, keep Whh cached
            float a0 = __builtin_nontemporal_load(&A[(size_t)t * DIM + j0]);
            float a1 = __builtin_nontemporal_load(&A[(size_t)t * DIM + j0 + 1]);
            out[(size_t)t * DIM + j0]     = tanhf(a0 + t0);
            out[(size_t)t * DIM + j0 + 1] = tanhf(a1 + t1);
        }

        // ---- device-wide barrier (monotonic counter in L2) ----
        __threadfence();          // push this block's h_t stores to agent scope
        __syncthreads();
        if (tid == 0) {
            __hip_atomic_fetch_add(bar, 1u, __ATOMIC_RELEASE,
                                   __HIP_MEMORY_SCOPE_AGENT);
            target += NBLK;
            while (__hip_atomic_load(bar, __ATOMIC_ACQUIRE,
                                     __HIP_MEMORY_SCOPE_AGENT) < target) {
                __builtin_amdgcn_s_sleep(1);
            }
        }
        __syncthreads();
        __threadfence();          // acquire side: see other blocks' h_t
    }
}

// ----------------------------------------------------------------------------
extern "C" void kernel_launch(void* const* d_in, const int* in_sizes, int n_in,
                              void* d_out, int out_size, void* d_ws, size_t ws_size,
                              hipStream_t stream) {
    const float* xs  = (const float*)d_in[0];   // (T, D)
    const float* wxh = (const float*)d_in[1];   // (D, D)
    const float* whh = (const float*)d_in[2];   // (D, D)
    float* out = (float*)d_out;                 // (T, 1, D) fp32

    // workspace layout (all 16B-aligned):
    //   A       : T*D fp32            32 MB
    //   whb     : D*D bf16             8 MB
    //   xsb     : T*D bf16            16 MB
    //   wxb     : D*D bf16             8 MB
    //   barrier : 1 u32
    char* ws = (char*)d_ws;
    float*  A   = (float*)ws;
    size_t  off = (size_t)T_STEPS * DIM * sizeof(float);
    __bf16* whb = (__bf16*)(ws + off);  off += (size_t)DIM * DIM * 2;
    __bf16* xsb = (__bf16*)(ws + off);  off += (size_t)T_STEPS * DIM * 2;
    __bf16* wxb = (__bf16*)(ws + off);  off += (size_t)DIM * DIM * 2;
    unsigned int* bar = (unsigned int*)(ws + off);

    init_barrier_kernel<<<1, 1, 0, stream>>>(bar);
    convert_bf16_kernel<<<2048, 256, 0, stream>>>(xs, wxh, whh, xsb, wxb, whb);
    // (T/16) x (D/128) = 256 x 16 = 4096 blocks, one 16x128 strip each
    gemm_xs_wxh_wmma<<<4096, 256, 0, stream>>>(xsb, wxb, A);
    rnn_recurrence<<<NBLK, 256, 0, stream>>>(A, whb, out, bar);
}
